// OptimizedInvariantMACE_5738076308128
// MI455X (gfx1250) — compile-verified, matrix operand order checked by reading, not run on hardware
//
#include <hip/hip_runtime.h>
#include <hip/hip_bf16.h>
#include <math.h>

// ---------------- problem constants (from reference) ----------------
#define NATOM 4000
#define NEDGE 100000
#define NGRAPH 32
#define NZ 10
#define NC 128
#define NK 16
#define NBESSEL 8
#define RMAX 5.0f
#define AVG_NEIGH 25.0f
#define INV_AVG (1.0f / AVG_NEIGH)
#define INV_SQRT_Z 0.31622776601683794f   // 1/sqrt(10)

typedef __attribute__((ext_vector_type(16))) __bf16 v16bf;
typedef __attribute__((ext_vector_type(2)))  __bf16 bf16x2;
typedef __attribute__((ext_vector_type(8)))  float v8f;

static __device__ __forceinline__ unsigned pack2bf(float a, float b) {
    bf16x2 t;
    t[0] = (__bf16)a;                  // native f32->bf16 (RNE), packs to cvt_pk
    t[1] = (__bf16)b;
    return __builtin_bit_cast(unsigned, t);
}
static __device__ __forceinline__ int lm_to_l(int k) {
    return (k < 1) ? 0 : (k < 4) ? 1 : (k < 9) ? 2 : 3;
}

// =====================================================================
// Unified GEMM body: 16x128 output tile, 256 threads (8 waves, one
// 16x16 D tile per wave), Kd = Nc = 128 fixed, K loop fully unrolled ->
// 4 v_wmma_f32_16x16x32_bf16. A is f32 (converted in flight, packed
// ds_store_b32); B is a pre-converted bf16 col-major tile Bt[col][k]
// staged with uint4 loads + ds_store_b128 (no conversion VALU).
// validRows is branchless (clamp + select) and constant-folds to
// nothing when the wrapper passes 16.
// =====================================================================
static __device__ __forceinline__ void gemm128_body(
    const float* __restrict__ A, long lda, int validRows,
    const __bf16* __restrict__ Bt,          // [128 cols][128 k] col-major bf16
    float* __restrict__ Co, long ldc,
    const float* __restrict__ addp, float scale,
    __bf16* As, unsigned* Bs32)             // As: 16*34 bf16; Bs32: 128*20 dwords
{
    const int tid  = threadIdx.x;           // 0..255
    const int lane = tid & 31;
    const int wave = tid >> 5;              // 0..7
    const int row  = lane & 15;
    const int half = lane >> 4;
    unsigned* As32 = (unsigned*)As;
    __bf16* Bs = (__bf16*)Bs32;             // stride 40 bf16 per col

    v8f acc;
#pragma unroll
    for (int i = 0; i < 8; ++i) acc[i] = 0.0f;

#pragma unroll
    for (int kk = 0; kk < NC; kk += 32) {
        __syncthreads();
        // A: 16 rows x 32 k = 256 float2's, one per thread (clamped rows)
        {
            int r  = tid >> 4;
            int kp = tid & 15;
            int rc = (r < validRows) ? r : (validRows - 1);
            float2 v = *reinterpret_cast<const float2*>(A + (long)rc * lda + kk + 2 * kp);
            if (r >= validRows) { v.x = 0.f; v.y = 0.f; }
            As32[r * 17 + kp] = pack2bf(v.x, v.y);
        }
        // B: copy 128 cols x 32 k of bf16; uint4 = 8 bf16 per load/store
#pragma unroll
        for (int t = 0; t < 2; ++t) {
            int g   = tid + t * 256;        // 0..511
            int col = g >> 2;               // 0..127
            int q   = g & 3;                // 4 x uint4 per col per k-step
            uint4 w = reinterpret_cast<const uint4*>(Bt + (long)col * NC + kk)[q];
            reinterpret_cast<uint4*>(Bs32 + col * 20)[q] = w;
        }
        __syncthreads();

        v16bf a, b;
#pragma unroll
        for (int e = 0; e < 16; ++e) {
            int k = half * 8 + e + ((e < 8) ? 0 : 8);
            a[e] = As[row * 34 + k];
            b[e] = Bs[(wave * 16 + row) * 40 + k];
        }
        acc = __builtin_amdgcn_wmma_f32_16x16x32_bf16(
                  false, a, false, b, (short)0, acc, false, false);
    }

    int col = wave * 16 + row;
#pragma unroll
    for (int r8 = 0; r8 < 8; ++r8) {
        int m = half * 8 + r8;
        if (m < validRows) {
            float v = acc[r8] * scale;
            if (addp) v += addp[(long)m * ldc + col];
            Co[(long)m * ldc + col] = v;
        }
    }
}

// plain row-major [M,128] @ bf16-Bt[128,128], M % 16 == 0
__global__ __launch_bounds__(256)
void gemm128_kernel(const float* __restrict__ A, const __bf16* __restrict__ Bt,
                    float* __restrict__ Co, const float* __restrict__ addp,
                    float scale)
{
    __shared__ __bf16 As[16 * 34];
    __shared__ unsigned Bs32[128 * 20];
    int rowBase = blockIdx.x * 16;
    gemm128_body(A + (long)rowBase * NC, NC, 16, Bt,
                 Co + (long)rowBase * NC, NC,
                 addp ? addp + (long)rowBase * NC : nullptr, scale, As, Bs32);
}

// per-k linear: msg2[n,k,:] = msg[n,k,:] @ lin_W[l(k)] / 25  (row stride K*C)
__global__ __launch_bounds__(256)
void lin_per_k_kernel(const float* __restrict__ msg, const __bf16* __restrict__ btLin,
                      float* __restrict__ msg2)
{
    __shared__ __bf16 As[16 * 34];
    __shared__ unsigned Bs32[128 * 20];
    int k = blockIdx.y;
    int l = lm_to_l(k);
    int rowBase = blockIdx.x * 16;
    gemm128_body(msg + (long)rowBase * NK * NC + k * NC, NK * NC, 16,
                 btLin + (long)l * NC * NC,
                 msg2 + (long)rowBase * NK * NC + k * NC, NK * NC,
                 nullptr, INV_AVG, As, Bs32);
}

// elemental skip: per-node, per-l C->C mix with skip_elem_W[z,l] (short rows)
__global__ __launch_bounds__(256)
void skip_elem_kernel(const float* __restrict__ msg2, const __bf16* __restrict__ btSkip,
                      const int* __restrict__ species, float* __restrict__ msg3)
{
    __shared__ __bf16 As[16 * 34];
    __shared__ unsigned Bs32[128 * 20];
    int n = blockIdx.x;
    int l = blockIdx.y;
    int z = species[n];
    gemm128_body(msg2 + (long)n * NK * NC + (long)(l * l) * NC, NC, 2 * l + 1,
                 btSkip + ((long)z * 4 + l) * NC * NC,
                 msg3 + (long)n * NK * NC + (long)(l * l) * NC, NC,
                 nullptr, INV_SQRT_Z, As, Bs32);
}

// ------- weight prep: f32 [nmats][Kd,Nc] row-major -> bf16 [nmats][NcPad,Kd]
// col-major (zero-padded cols), done once; all GEMM blocks then reuse it -------
__global__ __launch_bounds__(256)
void convert_bt_kernel(const float* __restrict__ src, __bf16* __restrict__ dst,
                       int Kd, int Nc, int NcPad, int nmats)
{
    long idx = (long)blockIdx.x * blockDim.x + threadIdx.x;
    long per = (long)NcPad * Kd;
    if (idx >= per * nmats) return;
    int mat = (int)(idx / per);
    long rem = idx % per;
    int col = (int)(rem / Kd);
    int k = (int)(rem % Kd);
    const float* s = src + (long)mat * Kd * Nc;
    dst[idx] = (col < Nc) ? (__bf16)s[(long)k * Nc + col] : (__bf16)0.0f;
}

// ---------------- edge geometry: SH (l<=3) + Bessel basis ----------------
__global__ __launch_bounds__(256)
void edge_geom_kernel(const float* __restrict__ pos, const float* __restrict__ shifts,
                      const int* __restrict__ edge, float* __restrict__ sh,
                      float* __restrict__ bes)
{
    int e = blockIdx.x * blockDim.x + threadIdx.x;
    if (e >= NEDGE) return;
    int recv = edge[e];
    int send = edge[NEDGE + e];
    float vx = pos[recv * 3 + 0] - pos[send * 3 + 0] + shifts[e * 3 + 0];
    float vy = pos[recv * 3 + 1] - pos[send * 3 + 1] + shifts[e * 3 + 1];
    float vz = pos[recv * 3 + 2] - pos[send * 3 + 2] + shifts[e * 3 + 2];
    float r = sqrtf(vx * vx + vy * vy + vz * vz);
    float inv = 1.0f / fmaxf(r, 1e-9f);
    float x = vx * inv, y = vy * inv, z = vz * inv;
    float x2 = x * x, y2 = y * y, z2 = z * z;
    float* S = sh + (long)e * NK;
    S[0]  = 1.0f;
    S[1]  = 1.7320508075688772f * y;
    S[2]  = 1.7320508075688772f * z;
    S[3]  = 1.7320508075688772f * x;
    S[4]  = 3.872983346207417f * x * y;
    S[5]  = 3.872983346207417f * y * z;
    S[6]  = 0.5f * 2.23606797749979f * (3.0f * z2 - 1.0f);
    S[7]  = 3.872983346207417f * x * z;
    S[8]  = 0.5f * 3.872983346207417f * (x2 - y2);
    S[9]  = 0.25f * 8.366600265340756f * y * (3.0f * x2 - y2);
    S[10] = 10.246950765959598f * x * y * z;
    S[11] = 0.25f * 6.48074069840786f * y * (5.0f * z2 - 1.0f);
    S[12] = 0.5f * 2.6457513110645907f * (5.0f * z2 - 3.0f) * z;
    S[13] = 0.25f * 6.48074069840786f * x * (5.0f * z2 - 1.0f);
    S[14] = 0.5f * 10.246950765959598f * z * (x2 - y2);
    S[15] = 0.25f * 8.366600265340756f * x * (x2 - y2);

    float rr = fmaxf(r, 1e-6f);
    float fc = 0.5f * (cosf((float)M_PI * fminf(r, RMAX) / RMAX) + 1.0f);
    float amp = sqrtf(2.0f / RMAX) / rr * fc;
    float* B = bes + (long)e * NBESSEL;
#pragma unroll
    for (int nb = 0; nb < NBESSEL; ++nb)
        B[nb] = amp * sinf((float)(nb + 1) * (float)M_PI * rr / RMAX);
}

// ---------------- node embedding gather ----------------
__global__ __launch_bounds__(256)
void embed_kernel(const float* __restrict__ embW, const int* __restrict__ species,
                  float* __restrict__ feats)
{
    int idx = blockIdx.x * blockDim.x + threadIdx.x;
    if (idx >= NATOM * NC) return;
    int n = idx >> 7, c = idx & 127;
    feats[idx] = embW[species[n] * NC + c];
}

// -------- fused message kernel: radial TP + SH outer product + scatter --------
__global__ __launch_bounds__(128)
void message_kernel(const float* __restrict__ f, const float* __restrict__ sh,
                    const float* __restrict__ bes, const float* __restrict__ radW,
                    const int* __restrict__ edge, float* __restrict__ msg)
{
    __shared__ float bsh[NBESSEL];
    __shared__ float ssh[NK];
    int e = blockIdx.x;
    int c = threadIdx.x;
    if (c < NBESSEL) bsh[c] = bes[(long)e * NBESSEL + c];
    if (c < NK)      ssh[c] = sh[(long)e * NK + c];
    __syncthreads();
    int recv = edge[e];
    int send = edge[NEDGE + e];
    float fs = f[(long)send * NC + c];
    float ef[4] = {0.f, 0.f, 0.f, 0.f};
#pragma unroll
    for (int nb = 0; nb < NBESSEL; ++nb) {
        float b = bsh[nb];
        const float* Wr = radW + (long)nb * (4 * NC) + c;
#pragma unroll
        for (int l = 0; l < 4; ++l) ef[l] = fmaf(b, Wr[l * NC], ef[l]);
    }
    float* dst = msg + (long)recv * NK * NC + c;
#pragma unroll
    for (int k = 0; k < NK; ++k) {
        float v = fs * ssh[k] * ef[lm_to_l(k)];
        atomicAdd(dst + k * NC, v);
    }
}

// ---------------- per-species contraction-tensor precompute ----------------
__global__ __launch_bounds__(256)
void u3w_kernel(const float* __restrict__ U3, const float* __restrict__ W3,
                float* __restrict__ U3W)
{
    long idx = (long)blockIdx.x * blockDim.x + threadIdx.x;
    if (idx >= (long)NZ * 4096 * NC) return;
    int c = (int)(idx % NC);
    long t = idx / NC;
    int ijk = (int)(t % 4096);
    int z = (int)(t / 4096);
    float s = 0.f;
#pragma unroll
    for (int p = 0; p < 8; ++p)
        s = fmaf(U3[ijk * 8 + p], W3[(z * 8 + p) * NC + c], s);
    U3W[idx] = s;
}

__global__ __launch_bounds__(256)
void u2w_kernel(const float* __restrict__ U2, const float* __restrict__ W2,
                float* __restrict__ U2W)
{
    long idx = (long)blockIdx.x * blockDim.x + threadIdx.x;
    if (idx >= (long)NZ * 256 * NC) return;
    int c = (int)(idx % NC);
    long t = idx / NC;
    int ij = (int)(t % 256);
    int z = (int)(t / 256);
    float s = 0.f;
#pragma unroll
    for (int p = 0; p < 4; ++p)
        s = fmaf(U2[ij * 4 + p], W2[(z * 4 + p) * NC + c], s);
    U2W[idx] = s;
}

__global__ __launch_bounds__(256)
void u1w_kernel(const float* __restrict__ U1, const float* __restrict__ W1,
                float* __restrict__ U1W)
{
    int idx = blockIdx.x * blockDim.x + threadIdx.x;
    if (idx >= NZ * NK * NC) return;
    int c = idx % NC;
    int t = idx / NC;
    int i = t % NK;
    int z = t / NK;
    U1W[idx] = U1[i] * W1[z * NC + c];
}

// ---------------- symmetric contraction (corr 3+2+1) ----------------
__global__ __launch_bounds__(128)
void symm_kernel(const float* __restrict__ msgin, const int* __restrict__ species,
                 const float* __restrict__ U3W, const float* __restrict__ U2W,
                 const float* __restrict__ U1W, float* __restrict__ hout)
{
    int n = blockIdx.x;
    int c = threadIdx.x;
    int z = species[n];
    float x[NK];
#pragma unroll
    for (int k = 0; k < NK; ++k) x[k] = msgin[(long)n * NK * NC + k * NC + c];

    const float* T3 = U3W + (long)z * 4096 * NC + c;
    const float* T2 = U2W + (long)z * 256 * NC + c;
    const float* T1 = U1W + (long)z * NK * NC + c;

    float res = 0.f;
    for (int i = 0; i < NK; ++i) {
        float oi = T1[i * NC];
        for (int j = 0; j < NK; ++j) {
            const float* row = T3 + (long)((i * 16 + j) * 16) * NC;
            __builtin_prefetch(row + 16 * NC, 0, 1);
            float t = T2[(i * 16 + j) * NC];
#pragma unroll
            for (int k = 0; k < NK; ++k)
                t = fmaf(row[k * NC], x[k], t);
            oi = fmaf(t, x[j], oi);
        }
        res = fmaf(oi, x[i], res);
    }
    hout[(long)n * NC + c] = res;
}

// ---------------- elemental FC skip: sc = h0 @ skip_fc_W[z] / sqrt(Z) --------
__global__ __launch_bounds__(256)
void skip_fc_kernel(const float* __restrict__ h0, const float* __restrict__ W,
                    const int* __restrict__ species, float* __restrict__ sc)
{
    int idx = blockIdx.x * blockDim.x + threadIdx.x;
    if (idx >= NATOM * NC) return;
    int n = idx >> 7, d = idx & 127;
    int z = species[n];
    const float* h = h0 + (long)n * NC;
    const float* Wz = W + (long)z * NC * NC + d;
    float s = 0.f;
#pragma unroll 8
    for (int c = 0; c < NC; ++c) s = fmaf(h[c], Wz[c * NC], s);
    sc[idx] = s * INV_SQRT_Z;
}

// ---------------- readouts + scale/shift + per-graph reduction ----------------
__global__ __launch_bounds__(128)
void final_kernel(const float* __restrict__ h0, const float* __restrict__ t16,
                  const float* __restrict__ r0W, const float* __restrict__ r2W,
                  const float* __restrict__ atomicE, const float* __restrict__ scale_p,
                  const float* __restrict__ shift_p, const int* __restrict__ species,
                  const int* __restrict__ batch, float* __restrict__ out)
{
    __shared__ float red[128];
    int n = blockIdx.x;
    int t = threadIdx.x;
    red[t] = h0[(long)n * NC + t] * r0W[t];
    __syncthreads();
    for (int s = 64; s > 0; s >>= 1) {
        if (t < s) red[t] += red[t + s];
        __syncthreads();
    }
    if (t == 0) {
        float es0 = red[0];
        float es1 = 0.f;
#pragma unroll
        for (int q = 0; q < 16; ++q) {
            float v = t16[(long)n * NC + q];        // t16 padded to 128 cols
            es1 += (v / (1.0f + expf(-v))) * r2W[q];
        }
        float node_e = scale_p[0] * (es0 + es1) + shift_p[0] + atomicE[species[n]];
        atomicAdd(&out[batch[n]], node_e);
    }
}

// =====================================================================
extern "C" void kernel_launch(void* const* d_in, const int* in_sizes, int n_in,
                              void* d_out, int out_size, void* d_ws, size_t ws_size,
                              hipStream_t stream)
{
    (void)in_sizes; (void)n_in; (void)out_size; (void)ws_size;
    const float* positions   = (const float*)d_in[0];
    const float* shifts      = (const float*)d_in[1];
    const float* emb_W       = (const float*)d_in[2];
    const float* atomic_e    = (const float*)d_in[3];
    const float* radial_W0   = (const float*)d_in[4];
    const float* lin_up_W0   = (const float*)d_in[5];
    const float* lin_W0      = (const float*)d_in[6];
    const float* skip_elem_W = (const float*)d_in[7];
    const float* radial_W1   = (const float*)d_in[8];
    const float* lin_up_W1   = (const float*)d_in[9];
    const float* lin_W1      = (const float*)d_in[10];
    const float* skip_fc_W   = (const float*)d_in[11];
    const float* U3          = (const float*)d_in[12];
    const float* U2          = (const float*)d_in[13];
    const float* U1          = (const float*)d_in[14];
    const float* W3_0        = (const float*)d_in[15];
    const float* W2_0        = (const float*)d_in[16];
    const float* W1_0        = (const float*)d_in[17];
    const float* W3_1        = (const float*)d_in[18];
    const float* W2_1        = (const float*)d_in[19];
    const float* W1_1        = (const float*)d_in[20];
    const float* prod_lin_W0 = (const float*)d_in[21];
    const float* prod_lin_W1 = (const float*)d_in[22];
    const float* readout0_W  = (const float*)d_in[23];
    const float* readout1_W1 = (const float*)d_in[24];
    const float* readout1_W2 = (const float*)d_in[25];
    const float* scale_p     = (const float*)d_in[26];
    const float* shift_p     = (const float*)d_in[27];
    const int*   species     = (const int*)d_in[28];
    const int*   edge_index  = (const int*)d_in[29];
    const int*   batch       = (const int*)d_in[30];
    float* out = (float*)d_out;

    // ---- workspace layout: f32 region then 16B-aligned bf16 weight cache ----
    float* W = (float*)d_ws;
    long off = 0;
    float* sh    = W + off; off += (long)NEDGE * NK;
    float* bes   = W + off; off += (long)NEDGE * NBESSEL;
    float* feats = W + off; off += (long)NATOM * NC;
    float* fbuf  = W + off; off += (long)NATOM * NC;
    float* msg   = W + off; off += (long)NATOM * NK * NC;
    float* msg2  = W + off; off += (long)NATOM * NK * NC;
    float* msg3  = W + off; off += (long)NATOM * NK * NC;
    float* u3w   = W + off; off += (long)NZ * 4096 * NC;
    float* u2w   = W + off; off += (long)NZ * 256 * NC;
    float* u1w   = W + off; off += (long)NZ * NK * NC;
    float* hpre  = W + off; off += (long)NATOM * NC;
    float* h0    = W + off; off += (long)NATOM * NC;
    float* sc    = W + off; off += (long)NATOM * NC;
    float* h1    = W + off; off += (long)NATOM * NC;
    float* t16   = W + off; off += (long)NATOM * NC;    // padded [N,128]
    off = (off + 3) & ~3L;                              // 16B align bf16 region

    __bf16* bt = (__bf16*)(W + off);
    const long MSZ = (long)NC * NC;                     // 16384 bf16 per matrix
    __bf16* bt_linup0 = bt;                 bt += MSZ;
    __bf16* bt_linup1 = bt;                 bt += MSZ;
    __bf16* bt_prod0  = bt;                 bt += MSZ;
    __bf16* bt_prod1  = bt;                 bt += MSZ;
    __bf16* bt_lin0   = bt;                 bt += 4 * MSZ;
    __bf16* bt_lin1   = bt;                 bt += 4 * MSZ;
    __bf16* bt_skip   = bt;                 bt += (long)NZ * 4 * MSZ;
    __bf16* bt_r1     = bt;                 bt += MSZ;

    const dim3 blk128(128), blk256(256);
    const dim3 gFast(NATOM / 16);                 // 250 blocks, 16x128 tiles
    const dim3 gLin(NATOM / 16, NK);              // 250 x 16
    const dim3 gSkip(NATOM, 4);                   // per node, per l
    const int cvtBlocks1  = (int)((MSZ + 255) / 256);
    const int cvtBlocks4  = (int)((4 * MSZ + 255) / 256);
    const int cvtBlocks40 = (int)((40 * MSZ + 255) / 256);

    // ---- one-time weight conversion to transposed bf16 ----
    convert_bt_kernel<<<cvtBlocks1, blk256, 0, stream>>>(lin_up_W0, bt_linup0, NC, NC, NC, 1);
    convert_bt_kernel<<<cvtBlocks1, blk256, 0, stream>>>(lin_up_W1, bt_linup1, NC, NC, NC, 1);
    convert_bt_kernel<<<cvtBlocks1, blk256, 0, stream>>>(prod_lin_W0, bt_prod0, NC, NC, NC, 1);
    convert_bt_kernel<<<cvtBlocks1, blk256, 0, stream>>>(prod_lin_W1, bt_prod1, NC, NC, NC, 1);
    convert_bt_kernel<<<cvtBlocks4, blk256, 0, stream>>>(lin_W0, bt_lin0, NC, NC, NC, 4);
    convert_bt_kernel<<<cvtBlocks4, blk256, 0, stream>>>(lin_W1, bt_lin1, NC, NC, NC, 4);
    convert_bt_kernel<<<cvtBlocks40, blk256, 0, stream>>>(skip_elem_W, bt_skip, NC, NC, NC, NZ * 4);
    convert_bt_kernel<<<cvtBlocks1, blk256, 0, stream>>>(readout1_W1, bt_r1, NC, 16, NC, 1);

    // geometry + embedding
    edge_geom_kernel<<<(NEDGE + 255) / 256, blk256, 0, stream>>>(positions, shifts, edge_index, sh, bes);
    embed_kernel<<<(NATOM * NC + 255) / 256, blk256, 0, stream>>>(emb_W, species, feats);

    // ======== layer 0 ========
    gemm128_kernel<<<gFast, blk256, 0, stream>>>(feats, bt_linup0, fbuf, nullptr, 1.0f);
    hipMemsetAsync(msg, 0, sizeof(float) * (size_t)NATOM * NK * NC, stream);
    message_kernel<<<NEDGE, blk128, 0, stream>>>(fbuf, sh, bes, radial_W0, edge_index, msg);
    lin_per_k_kernel<<<gLin, blk256, 0, stream>>>(msg, bt_lin0, msg2);
    skip_elem_kernel<<<gSkip, blk256, 0, stream>>>(msg2, bt_skip, species, msg3);
    u3w_kernel<<<(int)(((long)NZ * 4096 * NC + 255) / 256), blk256, 0, stream>>>(U3, W3_0, u3w);
    u2w_kernel<<<(NZ * 256 * NC + 255) / 256, blk256, 0, stream>>>(U2, W2_0, u2w);
    u1w_kernel<<<(NZ * NK * NC + 255) / 256, blk256, 0, stream>>>(U1, W1_0, u1w);
    symm_kernel<<<NATOM, blk128, 0, stream>>>(msg3, species, u3w, u2w, u1w, hpre);
    gemm128_kernel<<<gFast, blk256, 0, stream>>>(hpre, bt_prod0, h0, nullptr, 1.0f);
    skip_fc_kernel<<<(NATOM * NC + 255) / 256, blk256, 0, stream>>>(h0, skip_fc_W, species, sc);

    // ======== layer 1 ========
    gemm128_kernel<<<gFast, blk256, 0, stream>>>(h0, bt_linup1, fbuf, nullptr, 1.0f);
    hipMemsetAsync(msg, 0, sizeof(float) * (size_t)NATOM * NK * NC, stream);
    message_kernel<<<NEDGE, blk128, 0, stream>>>(fbuf, sh, bes, radial_W1, edge_index, msg);
    lin_per_k_kernel<<<gLin, blk256, 0, stream>>>(msg, bt_lin1, msg2);
    u3w_kernel<<<(int)(((long)NZ * 4096 * NC + 255) / 256), blk256, 0, stream>>>(U3, W3_1, u3w);
    u2w_kernel<<<(NZ * 256 * NC + 255) / 256, blk256, 0, stream>>>(U2, W2_1, u2w);
    u1w_kernel<<<(NZ * NK * NC + 255) / 256, blk256, 0, stream>>>(U1, W1_1, u1w);
    symm_kernel<<<NATOM, blk128, 0, stream>>>(msg2, species, u3w, u2w, u1w, hpre);
    gemm128_kernel<<<gFast, blk256, 0, stream>>>(hpre, bt_prod1, h1, sc, 1.0f);
    gemm128_kernel<<<gFast, blk256, 0, stream>>>(h1, bt_r1, t16, nullptr, 1.0f);

    // ======== readouts + graph reduction ========
    hipMemsetAsync(out, 0, sizeof(float) * NGRAPH, stream);
    final_kernel<<<NATOM, blk128, 0, stream>>>(h0, t16, readout0_W, readout1_W2,
                                               atomic_e, scale_p, shift_p,
                                               species, batch, out);
}
